// LinEq2v2_85478439125339
// MI455X (gfx1250) — compile-verified, matrix operand order checked by reading, not run on hardware
//
#include <hip/hip_runtime.h>

typedef __attribute__((ext_vector_type(16))) _Float16 v16h;
typedef __attribute__((ext_vector_type(8)))  float    v8f;

#define B_ 8
#define N_ 128
#define L_ 32
#define F_ 48

// workspace layout in float units
#define WS_ROWSUM 0
#define WS_COLSUM (WS_ROWSUM + B_*N_*L_)            // 32768
#define WS_DIAG   (WS_COLSUM + B_*N_*L_)            // 65536
#define WS_TRACE  (WS_DIAG   + B_*N_*L_)            // 98304
#define WS_TOTSUM (WS_TRACE  + B_*L_)               // 98560
#define WS_AROW   (WS_TOTSUM + B_*L_)               // 98816
#define WS_ACOL   (WS_AROW   + B_*N_*F_)            // 147968
#define WS_AD     (WS_ACOL   + B_*N_*F_)            // 197120
#define WS_WPACK  (WS_AD     + B_*N_*F_)            // 246272 (halves live here: 2*3*32*16)

// ---------------- Stage 1a: rowsum + diag per (b,i) ----------------
__global__ void __launch_bounds__(256) k_rowsum_diag(const float* __restrict__ x,
                                                     float* __restrict__ ws) {
    int bi  = blockIdx.x;              // b*N + i
    int tid = threadIdx.x;
    int l = tid & 31, jg = tid >> 5;   // 8 j-groups
    const float* base = x + (size_t)bi * N_ * L_;
    float acc = 0.f;
    for (int j = jg; j < N_; j += 8) acc += base[(size_t)j * L_ + l];
    __shared__ float red[8][32];
    red[jg][l] = acc;
    __syncthreads();
    if (jg == 0) {
        float t = 0.f;
        #pragma unroll
        for (int g = 0; g < 8; ++g) t += red[g][l];
        ws[WS_ROWSUM + (size_t)bi * L_ + l] = t;
        int i = bi & (N_ - 1);
        ws[WS_DIAG + (size_t)bi * L_ + l] = base[(size_t)i * L_ + l];
    }
}

// ---------------- Stage 1b: colsum per (b,j) ----------------
__global__ void __launch_bounds__(256) k_colsum(const float* __restrict__ x,
                                                float* __restrict__ ws) {
    int bj = blockIdx.x;               // b*N + j
    int b = bj >> 7, j = bj & (N_ - 1);
    int tid = threadIdx.x;
    int l = tid & 31, ig = tid >> 5;
    const float* base = x + (size_t)b * N_ * N_ * L_ + (size_t)j * L_;
    float acc = 0.f;
    for (int i = ig; i < N_; i += 8) acc += base[(size_t)i * N_ * L_ + l];
    __shared__ float red[8][32];
    red[ig][l] = acc;
    __syncthreads();
    if (ig == 0) {
        float t = 0.f;
        #pragma unroll
        for (int g = 0; g < 8; ++g) t += red[g][l];
        ws[WS_COLSUM + (size_t)bj * L_ + l] = t;
    }
}

// ---------------- Stage 1c: trace + totsum per b ----------------
__global__ void __launch_bounds__(256) k_trace_totsum(float* __restrict__ ws) {
    int b = blockIdx.x;
    int tid = threadIdx.x;
    int l = tid & 31, ig = tid >> 5;
    float tr = 0.f, ts = 0.f;
    for (int i = ig; i < N_; i += 8) {
        tr += ws[WS_DIAG   + (size_t)(b * N_ + i) * L_ + l];
        ts += ws[WS_ROWSUM + (size_t)(b * N_ + i) * L_ + l];
    }
    __shared__ float r1[8][32], r2[8][32];
    r1[ig][l] = tr; r2[ig][l] = ts;
    __syncthreads();
    if (ig == 0) {
        float a = 0.f, c = 0.f;
        #pragma unroll
        for (int g = 0; g < 8; ++g) { a += r1[g][l]; c += r2[g][l]; }
        ws[WS_TRACE  + b * L_ + l] = a;
        ws[WS_TOTSUM + b * L_ + l] = c;
    }
}

// ---------------- Stage 2: bias vectors per (b,i) ----------------
// Arow folds in the uniform term C[b,f] (e=13,14).
__global__ void __launch_bounds__(64) k_bias(const float* __restrict__ w,
                                             float* __restrict__ ws) {
    int bi = blockIdx.x;               // b*N + i
    int b = bi >> 7;
    int f = threadIdx.x;
    if (f >= F_) return;
    float arow = 0.f, acol = 0.f, ad = 0.f;
    for (int l = 0; l < L_; ++l) {
        float d  = ws[WS_DIAG   + (size_t)bi * L_ + l];
        float rs = ws[WS_ROWSUM + (size_t)bi * L_ + l];
        float cs = ws[WS_COLSUM + (size_t)bi * L_ + l];
        float tr = ws[WS_TRACE  + b * L_ + l];
        float ts = ws[WS_TOTSUM + b * L_ + l];
        const float* wl = w + (size_t)l * 15 * F_ + f;
        arow += d * wl[8*F_] + rs * wl[9*F_] + cs * wl[10*F_]
              + tr * wl[13*F_] + ts * wl[14*F_];          // row term + uniform C
        acol += d * wl[5*F_] + rs * wl[6*F_] + cs * wl[7*F_];
        ad   += d * wl[0]    + rs * wl[1*F_] + cs * wl[2*F_]
              + tr * wl[3*F_] + ts * wl[4*F_];
    }
    ws[WS_AROW + (size_t)bi * F_ + f] = arow;
    ws[WS_ACOL + (size_t)bi * F_ + f] = acol;
    ws[WS_AD   + (size_t)bi * F_ + f] = ad;
}

// ---------------- Stage 3: pack w[.,12,.] and w[.,11,.] into FP16 WMMA-B layout ----------------
// B (16-bit, 32x16): lane n=lane&15; elems 0..7 -> K=kb..kb+7; elems 8..15 -> K=16+kb..; kb=(lane>=16)?8:0
__global__ void __launch_bounds__(192) k_pack_w(const float* __restrict__ w,
                                                float* __restrict__ ws) {
    _Float16* wp = (_Float16*)(ws + WS_WPACK);
    int t = threadIdx.x;               // 0..191
    int op   = t / 96;                 // 0 -> e=12 (direct), 1 -> e=11 (transpose)
    int rem  = t % 96;
    int ft   = rem / 32;               // f-tile 0..2
    int lane = rem % 32;
    int n  = lane & 15;
    int kb = (lane >> 4) << 3;
    int f  = ft * 16 + n;
    int e  = (op == 0) ? 12 : 11;
    _Float16* dst = wp + ((size_t)(op * 3 + ft) * 32 + lane) * 16;
    #pragma unroll
    for (int p = 0; p < 8; ++p) {
        dst[p]     = (_Float16)w[(size_t)(kb + p)      * 15 * F_ + e * F_ + f];
        dst[8 + p] = (_Float16)w[(size_t)(16 + kb + p) * 15 * F_ + e * F_ + f];
    }
}

// ---------------- Stage 4: main WMMA kernel ----------------
// One wave per 16(j) x 48(f) output tile at fixed (b,i).
__global__ void __launch_bounds__(32) k_main(const float* __restrict__ x,
                                             const float* __restrict__ ws,
                                             float* __restrict__ out) {
    const v16h* wb = (const v16h*)(ws + WS_WPACK);
    int tile = blockIdx.x;
    int jt = tile & 7;
    int i  = (tile >> 3) & (N_ - 1);
    int b  = tile >> 10;
    int lane = threadIdx.x;
    int m  = lane & 15;
    int kb = (lane >> 4) << 3;         // 0 or 8
    int j0 = jt * 16;

    // A1 row m: x[b, i, j0+m, :]   A2 row m: x[b, j0+m, i, :]
    const float* xr1 = x + ((size_t)(b * N_ + i) * N_ + (j0 + m)) * L_;
    const float* xr2 = x + ((size_t)b * N_ + (j0 + m)) * N_ * L_ + (size_t)i * L_;

    v16h a1, a2;
    #pragma unroll
    for (int p = 0; p < 8; ++p) {
        a1[p]     = (_Float16)xr1[kb + p];
        a1[8 + p] = (_Float16)xr1[16 + kb + p];
        a2[p]     = (_Float16)xr2[kb + p];
        a2[8 + p] = (_Float16)xr2[16 + kb + p];
    }

    v8f c[3] = {};
    #pragma unroll
    for (int ft = 0; ft < 3; ++ft) {
        v16h bw12 = wb[(0 * 3 + ft) * 32 + lane];
        v16h bw11 = wb[(1 * 3 + ft) * 32 + lane];
        c[ft] = __builtin_amdgcn_wmma_f32_16x16x32_f16(
            false, a1, false, bw12, (short)0, c[ft], false, false);
        c[ft] = __builtin_amdgcn_wmma_f32_16x16x32_f16(
            false, a2, false, bw11, (short)0, c[ft], false, false);
    }

    // epilogue: bias add + store. D: reg r -> row r + 8*(lane>=16), col = lane&15
    // Diagonal term folded branchlessly (v_cndmask) to keep EXEC uniform around WMMA.
    int n    = lane & 15;
    int roff = (lane >> 4) << 3;
    const float* Arow = ws + WS_AROW + (size_t)(b * N_ + i) * F_;
    const float* Acol = ws + WS_ACOL + (size_t)b * N_ * F_;
    const float* Ad   = ws + WS_AD   + (size_t)(b * N_ + i) * F_;
    #pragma unroll
    for (int ft = 0; ft < 3; ++ft) {
        int f = ft * 16 + n;
        float ar = Arow[f];
        float ad = Ad[f];
        #pragma unroll
        for (int r = 0; r < 8; ++r) {
            int j = j0 + r + roff;
            float v = c[ft][r] + ar + Acol[(size_t)j * F_ + f];
            v += (j == i) ? ad : 0.0f;     // uniform control flow: select, not branch
            out[((size_t)(b * N_ + i) * N_ + j) * F_ + f] = v;
        }
    }
}

extern "C" void kernel_launch(void* const* d_in, const int* in_sizes, int n_in,
                              void* d_out, int out_size, void* d_ws, size_t ws_size,
                              hipStream_t stream) {
    const float* x = (const float*)d_in[0];   // [8,128,128,32] f32
    const float* w = (const float*)d_in[1];   // [32,15,48]     f32
    float* out = (float*)d_out;               // [8,128,128,48] f32
    float* ws  = (float*)d_ws;

    k_rowsum_diag <<<B_ * N_, 256, 0, stream>>>(x, ws);
    k_colsum      <<<B_ * N_, 256, 0, stream>>>(x, ws);
    k_trace_totsum<<<B_,      256, 0, stream>>>(ws);
    k_bias        <<<B_ * N_,  64, 0, stream>>>(w, ws);
    k_pack_w      <<<1,       192, 0, stream>>>(w, ws);
    k_main        <<<B_ * N_ * (N_ / 16), 32, 0, stream>>>(x, ws, out);
    (void)in_sizes; (void)n_in; (void)out_size; (void)ws_size;
}
